// NeuralTextureField_5677946765464
// MI455X (gfx1250) — matrix operand order repaired and checked by reading, not compiled, not run on hardware
//
#include <hip/hip_runtime.h>
#include <hip/hip_bf16.h>
#include <math.h>

typedef __attribute__((ext_vector_type(16))) _Float16 v16h;
typedef __attribute__((ext_vector_type(8)))  float    v8f;

#define N_LEVELS 16
#define LOG2_T   19
#define TBL      (1u << LOG2_T)          // 524288 entries per level
#define M_TILE   128
#define MT_CNT   (M_TILE / 16)           // 8 M-tiles
#define HID      512
#define LNB      0.36967849629863745     // ln(4096/16)/15 (double)

// LDS layout (bytes)
#define ENC_OFF 0
#define H1_OFF  (M_TILE * 32 * 2)                 // 8192
#define H2_OFF  (H1_OFF + M_TILE * HID * 2)       // 8192 + 131072
#define SMEM_BYTES (H2_OFF + M_TILE * HID * 2)    // 270336 (< 320KB WGP LDS)

// ---------------------------------------------------------------------------
// Prep: convert fp32 weights -> f16, pre-transposed to column-major so WMMA
// B-fragments are contiguous 16B loads. w2 is padded from 3 to 16 columns.
// w0T: [512][32], w1T: [512][512], w2T: [16][512]
// ---------------------------------------------------------------------------
__global__ void __launch_bounds__(256)
prep_weights(const float* __restrict__ w0,
             const float* __restrict__ w1,
             const float* __restrict__ w2,
             _Float16* __restrict__ w0T,
             _Float16* __restrict__ w1T,
             _Float16* __restrict__ w2T) {
  int i = blockIdx.x * 256 + threadIdx.x;
  if (i < 512 * 32) {
    int n = i >> 5, k = i & 31;
    w0T[i] = (_Float16)w0[k * 512 + n];
  } else if (i < 512 * 32 + 512 * 512) {
    int j = i - 512 * 32;
    int n = j >> 9, k = j & 511;
    w1T[j] = (_Float16)w1[k * 512 + n];
  } else if (i < 512 * 32 + 512 * 512 + 16 * 512) {
    int j = i - (512 * 32 + 512 * 512);
    int n = j >> 9, k = j & 511;
    w2T[j] = (_Float16)((n < 3) ? w2[k * 3 + n] : 0.0f);
  }
}

// ---------------------------------------------------------------------------
// Fragment loaders matching CDNA5 16-bit WMMA VGPR layouts (ISA 05_wmma §7.12.2)
// A 16x32 (MxK): lane m=L&15, hi=L>>4; halves 0..7 = K 8*hi..8*hi+7,
//                halves 8..15 = K 16+8*hi..23+8*hi   -> two contig 16B chunks
// B 32x16 (KxN): lane n=L&15, hi=L>>4; halves 0..7 = K 8*hi..8*hi+7, 8..15 = K 16+8*hi..23+8*hi
//                (column-major weight rows)          -> two contig 16B chunks
// ---------------------------------------------------------------------------
__device__ __forceinline__ v16h frag_a(const _Float16* base, int hi) {
  union { v16h v; uint4 q[2]; } u;
  u.q[0] = *(const uint4*)(base + 8 * hi);
  u.q[1] = *(const uint4*)(base + 16 + 8 * hi);
  return u.v;
}
__device__ __forceinline__ v16h frag_b(const _Float16* base, int hi) {
  union { v16h v; uint4 q[2]; } u;
  u.q[0] = *(const uint4*)(base + 8 * hi);
  u.q[1] = *(const uint4*)(base + 16 + 8 * hi);
  return u.v;
}

// ReLU + f16 convert + store a 16x16 f32 D-tile into LDS (row-major, stride HID).
// dst must point at H[(m0 + 8*hi)*HID + n0 + l15]; rows r are +r*HID halves.
__device__ __forceinline__ void store_tile_relu(_Float16* dst, const v8f& acc) {
  #pragma unroll
  for (int r = 0; r < 8; ++r) {
    float v = fmaxf(acc[r], 0.0f);          // single v_max_num_f32
    dst[r * HID] = (_Float16)v;             // ds_store_b16 base, offset:r*1024
  }
}

// ---------------------------------------------------------------------------
// Fused kernel: hash-grid encode (128 rows) -> 3 MLP layers via WMMA, all
// activations resident in LDS. 256 threads = 8 wave32.
// ---------------------------------------------------------------------------
__global__ void __launch_bounds__(256)
ntf_fused(const float* __restrict__ x,
          const float* __restrict__ table,
          const _Float16* __restrict__ w0T,
          const _Float16* __restrict__ w1T,
          const _Float16* __restrict__ w2T,
          float* __restrict__ out) {
  extern __shared__ char smem[];
  _Float16* enc = (_Float16*)(smem + ENC_OFF);   // [128][32] row-major f16
  _Float16* H1  = (_Float16*)(smem + H1_OFF);    // [128][512]
  _Float16* H2  = (_Float16*)(smem + H2_OFF);    // [128][512]

  const int tid  = threadIdx.x;
  const int wave = tid >> 5;
  const int lane = tid & 31;
  const int l15  = lane & 15;
  const int hi   = lane >> 4;
  const int row0 = blockIdx.x * M_TILE;

  // Warm L2/WGP$ with this wave's first w1T column block during encode.
  __builtin_prefetch(w1T + (wave * 16 + l15) * HID, 0, 1);

  // ---- Phase 0: hash-grid encoding (2048 tasks = 128 rows x 16 levels) ----
  #pragma unroll
  for (int i = 0; i < (M_TILE * N_LEVELS) / 256; ++i) {
    int task = tid + i * 256;
    int row  = task & (M_TILE - 1);
    int lvl  = task / M_TILE;
    // double precision so ceil() lands on the same side as the reference
    float scale = (float)(16.0 * exp((double)lvl * LNB) - 1.0);
    int res = (int)ceilf(scale) + 1;
    bool dense = (res * res) <= (int)TBL;
    float2 p = ((const float2*)x)[row0 + row];
    float px = p.x * scale + 0.5f, py = p.y * scale + 0.5f;
    float fx = floorf(px), fy = floorf(py);
    float wx = px - fx, wy = py - fy;
    unsigned ux = (unsigned)fx, uy = (unsigned)fy;
    const float* tab = table + (size_t)lvl * (size_t)TBL * 2;
    float o0 = 0.f, o1 = 0.f;
    #pragma unroll
    for (int c = 0; c < 4; ++c) {
      unsigned dx = c & 1, dy = c >> 1;
      unsigned cx = ux + dx, cy = uy + dy;
      unsigned idx = dense ? (cx + cy * (unsigned)res)
                           : ((cx ^ (cy * 2654435761u)) & (TBL - 1u));
      idx = idx < TBL ? idx : (TBL - 1u);
      float wgt = (dx ? wx : 1.f - wx) * (dy ? wy : 1.f - wy);
      o0 += tab[2 * idx]     * wgt;
      o1 += tab[2 * idx + 1] * wgt;
    }
    enc[row * 32 + 2 * lvl]     = (_Float16)o0;
    enc[row * 32 + 2 * lvl + 1] = (_Float16)o1;
  }
  __syncthreads();

  // ---- Layer 0: H1 = relu(enc @ W0), M=128 K=32 N=512 (1 K-step) ----
  for (int nt = wave; nt < 32; nt += 8) {
    v16h b = frag_b(w0T + (nt * 16 + l15) * 32, hi);
    _Float16* dcol = H1 + 8 * hi * HID + nt * 16 + l15;
    #pragma unroll
    for (int mt = 0; mt < MT_CNT; ++mt) {
      v16h a = frag_a(enc + (mt * 16 + l15) * 32, hi);
      v8f acc = {};
      acc = __builtin_amdgcn_wmma_f32_16x16x32_f16(false, a, false, b,
                                                   (short)0, acc, false, false);
      store_tile_relu(dcol + mt * 16 * HID, acc);
    }
  }
  __syncthreads();

  // ---- Layer 1: H2 = relu(H1 @ W1), M=128 K=512 N=512 ----
  for (int nt = wave; nt < 32; nt += 8) {
    const _Float16* bcol = w1T + (nt * 16 + l15) * HID;
    v8f acc[MT_CNT] = {};
    v16h b = frag_b(bcol, hi);                       // K-step 0
    for (int ks = 0; ks < 16; ++ks) {
      int k0 = ks * 32;
      // software pipeline: issue next B global load before this step's WMMAs
      v16h bn = (ks < 15) ? frag_b(bcol + k0 + 32, hi) : b;
      #pragma unroll
      for (int mt = 0; mt < MT_CNT; ++mt) {
        v16h a = frag_a(H1 + (mt * 16 + l15) * HID + k0, hi);
        acc[mt] = __builtin_amdgcn_wmma_f32_16x16x32_f16(false, a, false, b,
                                                         (short)0, acc[mt],
                                                         false, false);
      }
      b = bn;
    }
    _Float16* dcol = H2 + 8 * hi * HID + nt * 16 + l15;
    #pragma unroll
    for (int mt = 0; mt < MT_CNT; ++mt)
      store_tile_relu(dcol + mt * 16 * HID, acc[mt]);
  }
  __syncthreads();

  // ---- Layer 2: out = clip(H2 @ W2), N padded 3->16; one M-tile per wave ----
  {
    int mt = wave;                      // 8 waves, 8 M-tiles
    const _Float16* bcol = w2T + l15 * HID;
    v8f acc = {};
    v16h b = frag_b(bcol, hi);
    for (int ks = 0; ks < 16; ++ks) {
      int k0 = ks * 32;
      v16h bn = (ks < 15) ? frag_b(bcol + k0 + 32, hi) : b;
      v16h a = frag_a(H2 + (mt * 16 + l15) * HID + k0, hi);
      acc = __builtin_amdgcn_wmma_f32_16x16x32_f16(false, a, false, b,
                                                   (short)0, acc, false, false);
      b = bn;
    }
    if (l15 < 3) {
      float* drow = out + (row0 + mt * 16 + 8 * hi) * 3 + l15;
      #pragma unroll
      for (int r = 0; r < 8; ++r) {
        float v = fminf(fmaxf(acc[r], -0.95f), 0.95f);
        drow[r * 3] = v;
      }
    }
  }
}

// ---------------------------------------------------------------------------
extern "C" void kernel_launch(void* const* d_in, const int* in_sizes, int n_in,
                              void* d_out, int out_size, void* d_ws, size_t ws_size,
                              hipStream_t stream) {
  const float* x     = (const float*)d_in[0];
  const float* table = (const float*)d_in[1];
  const float* w0    = (const float*)d_in[2];
  const float* w1    = (const float*)d_in[3];
  const float* w2    = (const float*)d_in[4];
  float* out = (float*)d_out;
  const int N = in_sizes[0] / 2;

  _Float16* w0T = (_Float16*)d_ws;
  _Float16* w1T = (_Float16*)((char*)d_ws + 512 * 32 * 2);
  _Float16* w2T = (_Float16*)((char*)d_ws + 512 * 32 * 2 + 512 * 512 * 2);

  const int prepElems = 512 * 32 + 512 * 512 + 16 * 512;
  prep_weights<<<(prepElems + 255) / 256, 256, 0, stream>>>(w0, w1, w2,
                                                            w0T, w1T, w2T);

  const int nblocks = N / M_TILE;
  ntf_fused<<<nblocks, 256, SMEM_BYTES, stream>>>(x, table, w0T, w1T, w2T, out);
}
